// TriangleMultiplicationOutgoing_33681133535518
// MI455X (gfx1250) — matrix operand order repaired and checked by reading, not compiled, not run on hardware
//
#include <hip/hip_runtime.h>

// ---------------------------------------------------------------------------
// TriangleMultiplicationOutgoing for MI455X (gfx1250), wave32 + WMMA bf16.
// B=1, N=512, C=128 fixed by the reference.
// ---------------------------------------------------------------------------

typedef __bf16 bf16_t;
typedef __attribute__((ext_vector_type(16))) __bf16 v16bf;
typedef __attribute__((ext_vector_type(8)))  __bf16 v8bf;
typedef __attribute__((ext_vector_type(4)))  __bf16 v4bf;
typedef __attribute__((ext_vector_type(8)))  float  v8f;

constexpr int N = 512;
constexpr int C = 128;
constexpr int R = N * N;         // 262144 flattened (i,j) rows
constexpr int ZN_STRIDE = 136;   // 128 + 8 bf16 pad -> conflict-free frag reads
constexpr int LSTR = 40;         // 32 + 8 bf16 pad for k2 LDS panels

__device__ __forceinline__ v8f wmma_bf16(v16bf a, v16bf b, v8f c) {
  // D = A(16x32 bf16) x B(32x16 bf16) + C(16x16 f32)
  return __builtin_amdgcn_wmma_f32_16x16x32_bf16(
      /*neg_a=*/false, a, /*neg_b=*/false, b,
      /*c_mod=*/(short)0, c, /*reuse_a=*/false, /*reuse_b=*/false);
}

__device__ __forceinline__ v16bf join8(v8bf lo, v8bf hi) {
  v16bf r;
#pragma unroll
  for (int e = 0; e < 8; ++e) { r[e] = lo[e]; r[e + 8] = hi[e]; }
  return r;
}

// A fragment (16x32): lane holds row m = lane&15; elems 0..7 <-> K = half*8+e,
// elems 8..15 <-> K = 16 + half*8 + e  (ISA 7.12.2, 16-bit A 16x32)
__device__ __forceinline__ v16bf load_afrag(const bf16_t* rowp, int half) {
  return join8(*(const v8bf*)(rowp + half * 8),
               *(const v8bf*)(rowp + 16 + half * 8));
}

// B fragment (32x16): lane holds col n = lane&15; elems 0..7 <-> K = half*8+e,
// elems 8..15 <-> K = 16 + half*8 + e (same K order as A fragment).
// rowp points at the 32 contiguous K values belonging to column n.
__device__ __forceinline__ v16bf load_bfrag(const bf16_t* rowp, int half) {
  return join8(*(const v8bf*)(rowp + half * 8),
               *(const v8bf*)(rowp + 16 + half * 8));
}

__device__ __forceinline__ float wave_sum(float x) {
#pragma unroll
  for (int o = 16; o > 0; o >>= 1) x += __shfl_xor(x, o, 32);
  return x;
}

__device__ __forceinline__ float sigmoidf_(float x) {
  return 1.0f / (1.0f + __expf(-x));
}

// ---------------------------------------------------------------------------
// k0: convert the six (C,C) f32 weights to bf16, packed contiguously.
// ---------------------------------------------------------------------------
__global__ void k0_convert(const float* __restrict__ s0, const float* __restrict__ s1,
                           const float* __restrict__ s2, const float* __restrict__ s3,
                           const float* __restrict__ s4, const float* __restrict__ s5,
                           bf16_t* __restrict__ d) {
  int i = blockIdx.x * 256 + threadIdx.x;        // 6 * 16384 total
  int m = i >> 14, r = i & 16383;
  const float* s = (m == 0) ? s0 : (m == 1) ? s1 : (m == 2) ? s2
                 : (m == 3) ? s3 : (m == 4) ? s4 : s5;
  d[i] = (bf16_t)s[r];
}

// ---------------------------------------------------------------------------
// k1: LN(z) + 5 fused projections. One wave per 16-row tile, 8 waves/block.
// Writes a_t, b_t, g_t channel-major: buf[c*R + r], bf16.
// ---------------------------------------------------------------------------
__global__ void __launch_bounds__(256)
k1_ln_proj(const float* __restrict__ z, const float* __restrict__ mask,
           const float* __restrict__ lnw, const float* __restrict__ lnb,
           const bf16_t* __restrict__ Wpa, const bf16_t* __restrict__ Wpb,
           const bf16_t* __restrict__ Wga, const bf16_t* __restrict__ Wgb,
           const bf16_t* __restrict__ Wg,
           bf16_t* __restrict__ a_t, bf16_t* __restrict__ b_t,
           bf16_t* __restrict__ g_t) {
  __shared__ bf16_t zn[8][16 * ZN_STRIDE];
  const int wave = threadIdx.x >> 5, lane = threadIdx.x & 31;
  const int half = lane >> 4, ln = lane & 15;
  const size_t r0 = ((size_t)blockIdx.x * 8 + wave) * 16;

  const int c4 = lane * 4;
  const float4 wv = *(const float4*)(lnw + c4);
  const float4 bv = *(const float4*)(lnb + c4);

  // LayerNorm: 16 rows x 128 cols, 4 cols/lane, wave reduction for mu/var.
#pragma unroll 4
  for (int m = 0; m < 16; ++m) {
    float4 x = *(const float4*)(z + (r0 + m) * C + c4);
    float s  = x.x + x.y + x.z + x.w;
    float s2 = x.x * x.x + x.y * x.y + x.z * x.z + x.w * x.w;
    s = wave_sum(s); s2 = wave_sum(s2);
    float mu = s * (1.0f / C);
    float rstd = rsqrtf(s2 * (1.0f / C) - mu * mu + 1e-5f);
    v4bf v;
    v[0] = (bf16_t)((x.x - mu) * rstd * wv.x + bv.x);
    v[1] = (bf16_t)((x.y - mu) * rstd * wv.y + bv.y);
    v[2] = (bf16_t)((x.z - mu) * rstd * wv.z + bv.z);
    v[3] = (bf16_t)((x.w - mu) * rstd * wv.w + bv.w);
    *(v4bf*)(&zn[wave][m * ZN_STRIDE + c4]) = v;
  }
  __syncthreads();

  const bf16_t* zrow = &zn[wave][ln * ZN_STRIDE];
  v16bf afr[4];
#pragma unroll
  for (int kt = 0; kt < 4; ++kt) afr[kt] = load_afrag(zrow + kt * 32, half);

  float mk[8];
#pragma unroll
  for (int v = 0; v < 8; ++v) mk[v] = mask[r0 + half * 8 + v];  // row M=v+half*8

#pragma unroll 1
  for (int ot = 0; ot < 8; ++ot) {
    v8f aPA = {}, aGA = {}, aPB = {}, aGB = {}, aG = {};
#pragma unroll
    for (int kt = 0; kt < 4; ++kt) {
      const int wo = (ot * 16 + ln) * C + kt * 32;  // W row-major (out,in)
      aPA = wmma_bf16(afr[kt], load_bfrag(Wpa + wo, half), aPA);
      aGA = wmma_bf16(afr[kt], load_bfrag(Wga + wo, half), aGA);
      aPB = wmma_bf16(afr[kt], load_bfrag(Wpb + wo, half), aPB);
      aGB = wmma_bf16(afr[kt], load_bfrag(Wgb + wo, half), aGB);
      aG  = wmma_bf16(afr[kt], load_bfrag(Wg  + wo, half), aG);
    }
    v8bf av, bv2, gv;
#pragma unroll
    for (int v = 0; v < 8; ++v) {
      av[v]  = (bf16_t)(mk[v] * sigmoidf_(aGA[v]) * aPA[v]);
      bv2[v] = (bf16_t)(mk[v] * sigmoidf_(aGB[v]) * aPB[v]);
      gv[v]  = (bf16_t)sigmoidf_(aG[v]);
    }
    // D layout: lane col n=ln, vgpr v row M=v+half*8 -> 8 consecutive rows:
    // channel-major store is one contiguous 16-byte write per lane.
    const size_t base = (size_t)(ot * 16 + ln) * R + r0 + half * 8;
    *(v8bf*)(a_t + base) = av;
    *(v8bf*)(b_t + base) = bv2;
    *(v8bf*)(g_t + base) = gv;
  }
}

// ---------------------------------------------------------------------------
// k2: o[i,j,c] = sum_k a[i,k,c]*b[j,k,c].  Per block: one channel, 128x128
// (i x j) tile, 512 threads = 16 waves.  Each k-step stages 128x32 panels of
// A and B in padded LDS; each wave owns a 2x2 grid of 16x16 output tiles:
// 2 A-frags + 2 B-frags -> 4 WMMAs per barrier pair.
// ---------------------------------------------------------------------------
__global__ void __launch_bounds__(512)
k2_triangle(const bf16_t* __restrict__ a_t, const bf16_t* __restrict__ b_t,
            float* __restrict__ o) {
  __shared__ bf16_t lA[128 * LSTR];
  __shared__ bf16_t lB[128 * LSTR];
  const int c  = blockIdx.z;
  const int i0 = blockIdx.y * 128;
  const int j0 = blockIdx.x * 128;
  const int t = threadIdx.x, wave = t >> 5, lane = t & 31;
  const int half = lane >> 4, ln = lane & 15;
  const int iw = (wave & 3) * 2;       // first of two 16-row (i) subtiles
  const int jw = (wave >> 2) * 2;      // first of two 16-col (j) subtiles

  const bf16_t* Ab = a_t + (size_t)c * R + (size_t)i0 * N;
  const bf16_t* Bb = b_t + (size_t)c * R + (size_t)j0 * N;
  const int lrow = t >> 2;             // 0..127
  const int lcol = (t & 3) * 8;        // 0,8,16,24

  v8f acc00 = {}, acc01 = {}, acc10 = {}, acc11 = {};
#pragma unroll 1
  for (int k0 = 0; k0 < N; k0 += 32) {
    v8bf va = *(const v8bf*)(Ab + lrow * N + k0 + lcol);
    v8bf vb = *(const v8bf*)(Bb + lrow * N + k0 + lcol);
    __builtin_prefetch(Ab + lrow * N + k0 + 32 + lcol, 0, 1);  // global_prefetch
    __builtin_prefetch(Bb + lrow * N + k0 + 32 + lcol, 0, 1);
    __syncthreads();                    // previous iter's frag reads done
    *(v8bf*)(lA + lrow * LSTR + lcol) = va;
    *(v8bf*)(lB + lrow * LSTR + lcol) = vb;
    __syncthreads();
    v16bf af0 = load_afrag(lA + ((iw + 0) * 16 + ln) * LSTR, half);
    v16bf af1 = load_afrag(lA + ((iw + 1) * 16 + ln) * LSTR, half);
    v16bf bf0 = load_bfrag(lB + ((jw + 0) * 16 + ln) * LSTR, half);
    v16bf bf1 = load_bfrag(lB + ((jw + 1) * 16 + ln) * LSTR, half);
    acc00 = wmma_bf16(af0, bf0, acc00);
    acc01 = wmma_bf16(af0, bf1, acc01);
    acc10 = wmma_bf16(af1, bf0, acc10);
    acc11 = wmma_bf16(af1, bf1, acc11);
  }
#pragma unroll
  for (int v = 0; v < 8; ++v) {
    const size_t row0 = (size_t)(i0 + (iw + 0) * 16 + half * 8 + v) * N;
    const size_t row1 = (size_t)(i0 + (iw + 1) * 16 + half * 8 + v) * N;
    const size_t col0 = j0 + (jw + 0) * 16 + ln;
    const size_t col1 = j0 + (jw + 1) * 16 + ln;
    o[(row0 + col0) * C + c] = acc00[v];
    o[(row0 + col1) * C + c] = acc01[v];
    o[(row1 + col0) * C + c] = acc10[v];
    o[(row1 + col1) * C + c] = acc11[v];
  }
}

// ---------------------------------------------------------------------------
// k3: LN(o) + (on @ w_o.T) gated by precomputed g.  Same tiling as k1.
// ---------------------------------------------------------------------------
__global__ void __launch_bounds__(256)
k3_out(const float* __restrict__ o, const float* __restrict__ lnw,
       const float* __restrict__ lnb, const bf16_t* __restrict__ Wo,
       const bf16_t* __restrict__ g_t, float* __restrict__ out) {
  __shared__ bf16_t on[8][16 * ZN_STRIDE];
  const int wave = threadIdx.x >> 5, lane = threadIdx.x & 31;
  const int half = lane >> 4, ln = lane & 15;
  const size_t r0 = ((size_t)blockIdx.x * 8 + wave) * 16;
  const int c4 = lane * 4;
  const float4 wv = *(const float4*)(lnw + c4);
  const float4 bv = *(const float4*)(lnb + c4);

#pragma unroll 4
  for (int m = 0; m < 16; ++m) {
    float4 x = *(const float4*)(o + (r0 + m) * C + c4);
    float s  = x.x + x.y + x.z + x.w;
    float s2 = x.x * x.x + x.y * x.y + x.z * x.z + x.w * x.w;
    s = wave_sum(s); s2 = wave_sum(s2);
    float mu = s * (1.0f / C);
    float rstd = rsqrtf(s2 * (1.0f / C) - mu * mu + 1e-5f);
    v4bf v;
    v[0] = (bf16_t)((x.x - mu) * rstd * wv.x + bv.x);
    v[1] = (bf16_t)((x.y - mu) * rstd * wv.y + bv.y);
    v[2] = (bf16_t)((x.z - mu) * rstd * wv.z + bv.z);
    v[3] = (bf16_t)((x.w - mu) * rstd * wv.w + bv.w);
    *(v4bf*)(&on[wave][m * ZN_STRIDE + c4]) = v;
  }
  __syncthreads();

  const bf16_t* orow = &on[wave][ln * ZN_STRIDE];
  v16bf afr[4];
#pragma unroll
  for (int kt = 0; kt < 4; ++kt) afr[kt] = load_afrag(orow + kt * 32, half);

#pragma unroll 1
  for (int ot = 0; ot < 8; ++ot) {
    v8f acc = {};
#pragma unroll
    for (int kt = 0; kt < 4; ++kt)
      acc = wmma_bf16(afr[kt], load_bfrag(Wo + (ot * 16 + ln) * C + kt * 32, half), acc);
    const int cc = ot * 16 + ln;
    // g_t is channel-major; its layout matches the C/D fragment exactly.
    v8bf gv = *(const v8bf*)(g_t + (size_t)cc * R + r0 + half * 8);
#pragma unroll
    for (int v = 0; v < 8; ++v)
      out[(r0 + half * 8 + v) * C + cc] = (float)gv[v] * acc[v];
  }
}

// ---------------------------------------------------------------------------
extern "C" void kernel_launch(void* const* d_in, const int* in_sizes, int n_in,
                              void* d_out, int out_size, void* d_ws, size_t ws_size,
                              hipStream_t stream) {
  (void)in_sizes; (void)n_in; (void)out_size; (void)ws_size;
  const float* z    = (const float*)d_in[0];
  const float* mask = (const float*)d_in[1];
  const float* lniw = (const float*)d_in[2];
  const float* lnib = (const float*)d_in[3];
  const float* lnow = (const float*)d_in[4];
  const float* lnob = (const float*)d_in[5];
  const float* wpa  = (const float*)d_in[6];
  const float* wpb  = (const float*)d_in[7];
  const float* wga  = (const float*)d_in[8];
  const float* wgb  = (const float*)d_in[9];
  const float* wg   = (const float*)d_in[10];
  const float* wo   = (const float*)d_in[11];
  float* out = (float*)d_out;

  char* ws = (char*)d_ws;
  const size_t SZ_BF = (size_t)R * C * sizeof(bf16_t);   // 64 MB each
  bf16_t* a_t = (bf16_t*)(ws);
  bf16_t* b_t = (bf16_t*)(ws + SZ_BF);
  bf16_t* g_t = (bf16_t*)(ws + 2 * SZ_BF);
  float*  obuf = (float*)(ws + 3 * SZ_BF);               // 128 MB
  bf16_t* wbf = (bf16_t*)(ws + 3 * SZ_BF + (size_t)R * C * sizeof(float));
  bf16_t* Wpa = wbf + 0 * C * C;
  bf16_t* Wpb = wbf + 1 * C * C;
  bf16_t* Wga = wbf + 2 * C * C;
  bf16_t* Wgb = wbf + 3 * C * C;
  bf16_t* Wg  = wbf + 4 * C * C;
  bf16_t* Wo  = wbf + 5 * C * C;

  k0_convert<<<dim3(6 * C * C / 256), dim3(256), 0, stream>>>(
      wpa, wpb, wga, wgb, wg, wo, wbf);
  k1_ln_proj<<<dim3(R / 128), dim3(256), 0, stream>>>(
      z, mask, lniw, lnib, Wpa, Wpb, Wga, Wgb, Wg, a_t, b_t, g_t);
  k2_triangle<<<dim3(N / 128, N / 128, C), dim3(512), 0, stream>>>(a_t, b_t, obuf);
  k3_out<<<dim3(R / 128), dim3(256), 0, stream>>>(obuf, lnow, lnob, Wo, g_t, out);
}
